// OmniGenLayer_19198503813604
// MI455X (gfx1250) — compile-verified, matrix operand order checked by reading, not run on hardware
//
#include <hip/hip_runtime.h>

#define SEQ   2048
#define HID   2048
#define NH    32
#define HD    64
#define INTER 8192

typedef __attribute__((ext_vector_type(16))) __bf16 v16bf;
typedef __attribute__((ext_vector_type(8)))  float  v8f;
typedef unsigned int u32;
typedef __attribute__((ext_vector_type(4))) u32 u32x4;
typedef __attribute__((ext_vector_type(8))) u32 u32x8;
using bf16 = __bf16;

static __device__ __forceinline__ v8f wmma_bf16(v16bf a, v16bf b, v8f c) {
  // D = A(16x32 bf16) * B(32x16 bf16) + C(16x16 f32)
  return __builtin_amdgcn_wmma_f32_16x16x32_bf16(false, a, false, b, (short)0, c,
                                                 false, false);
}

// Load a 16x32 bf16 fragment (A-layout; B uses same layout when the source is
// stored column-major, i.e. lane = output column, K contiguous).
// Per ISA 7.12.2: lanes 0-15 row M=lane hold K{0..7,16..23};
// lanes 16-31 row M=lane-16 hold K{8..15,24..31}.
template <typename T>
static __device__ __forceinline__ v16bf load_frag(const T* base, int ld) {
  const int lane = threadIdx.x & 31;
  const int row  = lane & 15;
  const int half = lane >> 4;
  const T* p = base + row * ld + half * 8;
  union { v16bf v; uint4 q[2]; } u;
  u.q[0] = *reinterpret_cast<const uint4*>(p);
  u.q[1] = *reinterpret_cast<const uint4*>(p + 16);
  return u.v;
}

// ---------------- TDM: 2D bf16 tile DMA  global -> LDS -----------------------
// Builds D# group0 (4 SGPRs) + group1 (8 SGPRs) per ISA 08_async_tensor 8.3/8.4
// and issues tensor_load_to_lds (2-group form, <=2D tensor).
// Pads LDS rows: every 16 DWORDs (32 bf16) insert 4 DWORDs (8 bf16) -> ld 40.
static __device__ __forceinline__ void tdm_load_2d(const void* gaddr, u32 lds_off,
                                                   u32 tile_d0, u32 tile_d1,
                                                   u32 tensor_d0, u32 stride_d0) {
  unsigned long long ga = (unsigned long long)gaddr;
  const u32 tensor_d1 = 0x40000000u;  // effectively unbounded (in-bounds use)
  u32x4 g0;
  g0[0] = 1u;                                      // count=1, user descriptor
  g0[1] = lds_off;                                 // lds_addr (bytes)
  g0[2] = (u32)ga;                                 // global_addr[31:0]
  g0[3] = (u32)((ga >> 32) & 0x01FFFFFFull) | (2u << 30);  // ga[56:32], type=2
  u32x8 g1;
  g1[0] = (1u << 16)        // data_size = 1 (2 bytes, bf16)
        | (1u << 20)        // pad_enable
        | (3u << 22)        // pad_interval: 16 DWORDs
        | (3u << 25);       // pad_amount: 4 DWORDs
  g1[1] = (tensor_d0 & 0xFFFFu) << 16;                       // tensor_dim0 lo
  g1[2] = (tensor_d0 >> 16) | ((tensor_d1 & 0xFFFFu) << 16); // dim0 hi | dim1 lo
  g1[3] = (tensor_d1 >> 16) | (tile_d0 << 16);               // dim1 hi | tile0
  g1[4] = tile_d1 & 0xFFFFu;                                 // tile1, tile2=0
  g1[5] = stride_d0;                                         // dim0 stride lo
  g1[6] = 0u;                                                // stride hi, dim1s
  g1[7] = 0u;
  asm volatile("tensor_load_to_lds %0, %1" :: "s"(g0), "s"(g1) : "memory");
}

// ---------------- weight convert fp32[K,N] -> bf16 transposed [N,K] ----------
__global__ void k_cvt_transpose(const float* __restrict__ w, bf16* __restrict__ wT,
                                int K, int N) {
  long long id = (long long)blockIdx.x * blockDim.x + threadIdx.x;
  long long total = (long long)K * N;
  if (id >= total) return;
  int kk = (int)(id / N);
  int nn = (int)(id % N);
  wT[(long long)nn * K + kk] = (bf16)w[id];
}

// ---------------- LayerNorm: one block per row, 256 threads ------------------
__global__ __launch_bounds__(256)
void k_layernorm(const float* __restrict__ x, const float* __restrict__ gamma,
                 const float* __restrict__ beta, bf16* __restrict__ out) {
  const int row = blockIdx.x;
  const float* xr = x + (long long)row * HID;
  __shared__ float red0[8];
  __shared__ float red1[8];
  float v[8];
  float s = 0.f, s2 = 0.f;
#pragma unroll
  for (int i = 0; i < 8; ++i) {
    v[i] = xr[threadIdx.x + i * 256];
    s += v[i];
    s2 += v[i] * v[i];
  }
#pragma unroll
  for (int m = 1; m < 32; m <<= 1) {
    s  += __shfl_xor(s, m, 32);
    s2 += __shfl_xor(s2, m, 32);
  }
  const int w = threadIdx.x >> 5;
  if ((threadIdx.x & 31) == 0) { red0[w] = s; red1[w] = s2; }
  __syncthreads();
  float ts = 0.f, ts2 = 0.f;
#pragma unroll
  for (int i = 0; i < 8; ++i) { ts += red0[i]; ts2 += red1[i]; }
  const float mu = ts * (1.0f / HID);
  const float var = ts2 * (1.0f / HID) - mu * mu;
  const float rstd = rsqrtf(var + 1e-5f);
  bf16* outr = out + (long long)row * HID;
#pragma unroll
  for (int i = 0; i < 8; ++i) {
    int idx = threadIdx.x + i * 256;
    outr[idx] = (bf16)((v[i] - mu) * rstd * gamma[idx] + beta[idx]);
  }
}

// ---------------- bf16 WMMA GEMM: C[M,N] = A[M,K] * WT[N,K]^T ----------------
// Block tile 128x128, 8 waves as 2(M) x 4(N); wave tile 64x32 = 4x2 WMMA tiles.
// Operand staging: double-buffered Tensor Data Mover DMA; wave 0 issues the
// next K-tile's loads before waiting on the current pair (TENSORcnt in-order),
// overlapping DMA with the WMMA pipe.
#define BM 128
#define BN 128
#define BK 32
#define LDA 40
#define LDB 40

#define EPI_QKV     1
#define EPI_ORES    2
#define EPI_BIAS    3
#define EPI_BIASRES 4

template <int EPI>
__global__ __launch_bounds__(256)
void k_gemm(const bf16* __restrict__ A, const bf16* __restrict__ WT,
            int N, int K,
            float* __restrict__ outF, bf16* __restrict__ outB,
            const float* __restrict__ bias, const float* __restrict__ residual,
            bf16* __restrict__ oq, bf16* __restrict__ okk, bf16* __restrict__ ovT) {
  __shared__ bf16 sA[2][BM * LDA];
  __shared__ bf16 sB[2][BN * LDB];
  const int tid = threadIdx.x;
  const int bM = blockIdx.y;
  const int bN = blockIdx.x;
  const int wid = tid >> 5;
  const int wm = wid & 1;   // 2 waves in M
  const int wn = wid >> 1;  // 4 waves in N

  v8f zero = {};
  v8f acc[4][2];
#pragma unroll
  for (int i = 0; i < 4; ++i)
#pragma unroll
    for (int j = 0; j < 2; ++j) acc[i][j] = zero;

  const bf16* Ab = A + (long long)(bM * BM) * K;
  const bf16* Bb = WT + (long long)(bN * BN) * K;
  const u32 ldsA0 = (u32)(size_t)&sA[0][0];  // low 32 bits == LDS byte offset
  const u32 ldsA1 = (u32)(size_t)&sA[1][0];
  const u32 ldsB0 = (u32)(size_t)&sB[0][0];
  const u32 ldsB1 = (u32)(size_t)&sB[1][0];
  const int nk = K / BK;

  // prologue: stage K-tile 0 into buffer 0
  if (wid == 0) {
    tdm_load_2d(Ab, ldsA0, BK, BM, (u32)K, (u32)K);
    tdm_load_2d(Bb, ldsB0, BK, BN, (u32)K, (u32)K);
  }

  for (int kt = 0; kt < nk; ++kt) {
    const int cur = kt & 1;
    // barrier 1: destination buffer of the next DMA is no longer being read
    __syncthreads();
    if (wid == 0) {
      if (kt + 1 < nk) {
        tdm_load_2d(Ab + (kt + 1) * BK, cur ? ldsA0 : ldsA1, BK, BM, (u32)K, (u32)K);
        tdm_load_2d(Bb + (kt + 1) * BK, cur ? ldsB0 : ldsB1, BK, BN, (u32)K, (u32)K);
        __builtin_amdgcn_s_wait_tensorcnt(2);  // current pair retired, next in flight
      } else {
        __builtin_amdgcn_s_wait_tensorcnt(0);
      }
    }
    // barrier 2: publish current buffer to all waves
    __syncthreads();
    const bf16* cA = &sA[cur][0];
    const bf16* cB = &sB[cur][0];
    v16bf a[4], b[2];
#pragma unroll
    for (int i = 0; i < 4; ++i)
      a[i] = load_frag(cA + (wm * 64 + i * 16) * LDA, LDA);
#pragma unroll
    for (int j = 0; j < 2; ++j)
      b[j] = load_frag(cB + (wn * 32 + j * 16) * LDB, LDB);
#pragma unroll
    for (int i = 0; i < 4; ++i)
#pragma unroll
      for (int j = 0; j < 2; ++j)
        acc[i][j] = wmma_bf16(a[i], b[j], acc[i][j]);
  }

  const int lane = tid & 31;
  const int lrow = 8 * (lane >> 4);
  const int lcol = lane & 15;
#pragma unroll
  for (int mi = 0; mi < 4; ++mi)
#pragma unroll
    for (int ni = 0; ni < 2; ++ni)
#pragma unroll
      for (int e = 0; e < 8; ++e) {
        int row = bM * BM + wm * 64 + mi * 16 + lrow + e;
        int col = bN * BN + wn * 32 + ni * 16 + lcol;
        float va = acc[mi][ni][e];
        if (EPI == EPI_QKV) {
          int which = col >> 11;      // 0:q 1:k 2:v
          int rem = col & 2047;
          int h = rem >> 6, d = rem & 63;
          bf16 bv = (bf16)va;
          if (which == 0)      oq [((long long)(h * SEQ + row)) * HD + d] = bv;
          else if (which == 1) okk[((long long)(h * SEQ + row)) * HD + d] = bv;
          else                 ovT[((long long)(h * HD + d)) * SEQ + row] = bv;
        } else if (EPI == EPI_ORES) {
          outF[(long long)row * N + col] = va + residual[(long long)row * N + col];
        } else if (EPI == EPI_BIAS) {
          outB[(long long)row * N + col] = (bf16)(va + bias[col]);
        } else if (EPI == EPI_BIASRES) {
          outF[(long long)row * N + col] =
              va + bias[col] + residual[(long long)row * N + col];
        }
      }
}

// ---------------- Flash attention: wave = 16 q-rows, key tiles of 32 ---------
__global__ __launch_bounds__(256)
void k_attn(const bf16* __restrict__ q, const bf16* __restrict__ k,
            const bf16* __restrict__ vT, const float* __restrict__ mask,
            bf16* __restrict__ ctx) {
  __shared__ bf16 plds[8][16][32];
  const int tiles = SEQ / 128;
  const int h  = blockIdx.x / tiles;
  const int qt = blockIdx.x % tiles;
  const int wid  = threadIdx.x >> 5;
  const int lane = threadIdx.x & 31;
  const int half = lane >> 4;
  const int lcol = lane & 15;
  const int qrow0 = qt * 128 + wid * 16;
  const float scale = 0.125f;  // 1/sqrt(64)

  const bf16* qb = q + ((long long)(h * SEQ + qrow0)) * HD;
  v16bf aq0 = load_frag(qb, HD);        // d = 0..31
  v16bf aq1 = load_frag(qb + 32, HD);   // d = 32..63

  float m[8], l[8];
#pragma unroll
  for (int r = 0; r < 8; ++r) { m[r] = -3.0e38f; l[r] = 0.f; }
  v8f zero = {};
  v8f o[4];
#pragma unroll
  for (int t = 0; t < 4; ++t) o[t] = zero;

  for (int kb = 0; kb < SEQ; kb += 32) {
    v8f s0 = zero, s1 = zero;
    {
      const bf16* k0 = k + ((long long)(h * SEQ + kb)) * HD;
      v16bf b00 = load_frag(k0, HD);
      v16bf b01 = load_frag(k0 + 32, HD);
      v16bf b10 = load_frag(k0 + 16 * HD, HD);
      v16bf b11 = load_frag(k0 + 16 * HD + 32, HD);
      s0 = wmma_bf16(aq0, b00, s0);
      s0 = wmma_bf16(aq1, b01, s0);
      s1 = wmma_bf16(aq0, b10, s1);
      s1 = wmma_bf16(aq1, b11, s1);
    }
    const float mk0 = (1.0f - mask[kb + lcol]) * -10000.0f;
    const float mk1 = (1.0f - mask[kb + 16 + lcol]) * -10000.0f;
    float rm[8];
#pragma unroll
    for (int r = 0; r < 8; ++r) {
      float a = s0[r] * scale + mk0;
      float b = s1[r] * scale + mk1;
      s0[r] = a; s1[r] = b;
      rm[r] = fmaxf(a, b);
    }
#pragma unroll
    for (int xm = 1; xm < 16; xm <<= 1)
#pragma unroll
      for (int r = 0; r < 8; ++r)
        rm[r] = fmaxf(rm[r], __shfl_xor(rm[r], xm, 32));
    float alpha[8], rs[8];
#pragma unroll
    for (int r = 0; r < 8; ++r) {
      float mn = fmaxf(m[r], rm[r]);
      alpha[r] = __expf(m[r] - mn);
      m[r] = mn;
      float p0 = __expf(s0[r] - mn);
      float p1 = __expf(s1[r] - mn);
      s0[r] = p0; s1[r] = p1;
      rs[r] = p0 + p1;
    }
#pragma unroll
    for (int xm = 1; xm < 16; xm <<= 1)
#pragma unroll
      for (int r = 0; r < 8; ++r)
        rs[r] += __shfl_xor(rs[r], xm, 32);
#pragma unroll
    for (int r = 0; r < 8; ++r) l[r] = l[r] * alpha[r] + rs[r];
#pragma unroll
    for (int t = 0; t < 4; ++t)
#pragma unroll
      for (int e = 0; e < 8; ++e) o[t][e] *= alpha[e];
    // reshape P (C-layout f32) -> A-layout bf16 through per-wave LDS tile
#pragma unroll
    for (int r = 0; r < 8; ++r) {
      plds[wid][r + 8 * half][lcol]      = (bf16)s0[r];
      plds[wid][r + 8 * half][16 + lcol] = (bf16)s1[r];
    }
    v16bf pa = load_frag(&plds[wid][0][0], 32);
#pragma unroll
    for (int t = 0; t < 4; ++t) {
      v16bf bv = load_frag(vT + ((long long)(h * HD + t * 16)) * SEQ + kb, SEQ);
      o[t] = wmma_bf16(pa, bv, o[t]);
    }
  }
#pragma unroll
  for (int t = 0; t < 4; ++t)
#pragma unroll
    for (int e = 0; e < 8; ++e) {
      int row = qrow0 + e + 8 * half;
      int col = h * HD + t * 16 + lcol;
      ctx[(long long)row * HID + col] = (bf16)(o[t][e] / l[e]);
    }
}

// ---------------- SiLU(gate) * up -------------------------------------------
__global__ void k_silu_mul(const bf16* __restrict__ gu, bf16* __restrict__ act) {
  long long id = (long long)blockIdx.x * 256 + threadIdx.x;
  int srow = (int)(id / INTER);
  int i = (int)(id % INTER);
  float g = (float)gu[(long long)srow * (2 * INTER) + i];
  float u = (float)gu[(long long)srow * (2 * INTER) + INTER + i];
  act[id] = (bf16)(g / (1.f + __expf(-g)) * u);
}

// ---------------- driver -----------------------------------------------------
extern "C" void kernel_launch(void* const* d_in, const int* in_sizes, int n_in,
                              void* d_out, int out_size, void* d_ws, size_t ws_size,
                              hipStream_t stream) {
  const float* hs    = (const float*)d_in[0];
  const float* amask = (const float*)d_in[1];
  const float* g1    = (const float*)d_in[2];
  const float* b1    = (const float*)d_in[3];
  const float* wqkv  = (const float*)d_in[4];
  const float* wo    = (const float*)d_in[5];
  const float* g2    = (const float*)d_in[6];
  const float* b2    = (const float*)d_in[7];
  const float* wgu   = (const float*)d_in[8];
  const float* bgu   = (const float*)d_in[9];
  const float* wd    = (const float*)d_in[10];
  const float* bd    = (const float*)d_in[11];
  float* out = (float*)d_out;

  char* p = (char*)d_ws;
  auto alloc = [&](size_t bytes) {
    char* r = p;
    p += (bytes + 255) & ~(size_t)255;
    return r;
  };
  bf16* xb     = (bf16*)alloc((size_t)SEQ * HID * 2);
  bf16* wqkvT  = (bf16*)alloc((size_t)3 * HID * HID * 2);
  bf16* woT    = (bf16*)alloc((size_t)HID * HID * 2);
  bf16* wguT   = (bf16*)alloc((size_t)2 * INTER * HID * 2);
  bf16* wdT    = (bf16*)alloc((size_t)HID * INTER * 2);
  bf16* qbuf   = (bf16*)alloc((size_t)NH * SEQ * HD * 2);
  bf16* kbuf   = (bf16*)alloc((size_t)NH * SEQ * HD * 2);
  bf16* vTbuf  = (bf16*)alloc((size_t)NH * HD * SEQ * 2);
  bf16* ctx    = (bf16*)alloc((size_t)SEQ * HID * 2);
  float* hidden= (float*)alloc((size_t)SEQ * HID * 4);
  bf16* x2     = (bf16*)alloc((size_t)SEQ * HID * 2);
  bf16* gu     = (bf16*)alloc((size_t)SEQ * 2 * INTER * 2);
  bf16* act    = (bf16*)alloc((size_t)SEQ * INTER * 2);

  // 1) weights fp32 -> bf16 transposed
  {
    long long t;
    t = (long long)HID * (3 * HID);
    k_cvt_transpose<<<(t + 255) / 256, 256, 0, stream>>>(wqkv, wqkvT, HID, 3 * HID);
    t = (long long)HID * HID;
    k_cvt_transpose<<<(t + 255) / 256, 256, 0, stream>>>(wo, woT, HID, HID);
    t = (long long)HID * (2 * INTER);
    k_cvt_transpose<<<(t + 255) / 256, 256, 0, stream>>>(wgu, wguT, HID, 2 * INTER);
    t = (long long)INTER * HID;
    k_cvt_transpose<<<(t + 255) / 256, 256, 0, stream>>>(wd, wdT, INTER, HID);
  }
  // 2) LN1
  k_layernorm<<<SEQ, 256, 0, stream>>>(hs, g1, b1, xb);
  // 3) QKV GEMM (N=6144) with q/k/vT scatter
  k_gemm<EPI_QKV><<<dim3((3 * HID) / BN, SEQ / BM), 256, 0, stream>>>(
      xb, wqkvT, 3 * HID, HID, nullptr, nullptr, nullptr, nullptr,
      qbuf, kbuf, vTbuf);
  // 4) attention
  k_attn<<<NH * (SEQ / 128), 256, 0, stream>>>(qbuf, kbuf, vTbuf, amask, ctx);
  // 5) O projection + residual -> hidden (fp32)
  k_gemm<EPI_ORES><<<dim3(HID / BN, SEQ / BM), 256, 0, stream>>>(
      ctx, woT, HID, HID, hidden, nullptr, nullptr, hs, nullptr, nullptr, nullptr);
  // 6) LN2
  k_layernorm<<<SEQ, 256, 0, stream>>>(hidden, g2, b2, x2);
  // 7) gate_up GEMM (N=16384) + bias -> bf16
  k_gemm<EPI_BIAS><<<dim3((2 * INTER) / BN, SEQ / BM), 256, 0, stream>>>(
      x2, wguT, 2 * INTER, HID, nullptr, gu, bgu, nullptr, nullptr, nullptr, nullptr);
  // 8) silu(gate) * up
  {
    long long t = (long long)SEQ * INTER;
    k_silu_mul<<<(t + 255) / 256, 256, 0, stream>>>(gu, act);
  }
  // 9) down GEMM (K=8192) + bias + residual -> out (fp32)
  k_gemm<EPI_BIASRES><<<dim3(HID / BN, SEQ / BM), 256, 0, stream>>>(
      act, wdT, HID, INTER, out, nullptr, bd, hidden, nullptr, nullptr, nullptr);
}
